// _IterativeFeatureExclusion_39298950758607
// MI455X (gfx1250) — compile-verified, matrix operand order checked by reading, not run on hardware
//
#include <hip/hip_runtime.h>

// ---------------------------------------------------------------------------
// IterativeFeatureExclusion for MI455X (gfx1250, wave32, WMMA).
//
//   z[j,a,b,u]  = (X @ K1)[b,(j,a,u)] - x[b,j]*Kdiag[j,a,u]     (GEMM1 + rank-1 fix)
//   zs          = softmax_u(z)
//   y[b,f]      = zs[b,(j,a,u)] @ W2[(j,a,u),f],  W2 = exp(2K)*fc_w[a]/F
//   out         = softmax_f(BN(y))
//
// U padded 10->16 so each (j,a) group is one WMMA N-tile. Softmax is done in
// A-fragment layout (a row's u-values live in just 2 lanes) -> one shfl_xor
// per reduction instead of a 16-lane tree.
// ---------------------------------------------------------------------------

typedef __attribute__((ext_vector_type(16))) __bf16 v16bf;
typedef __attribute__((ext_vector_type(8)))  __bf16 v8bf;
typedef __attribute__((ext_vector_type(8)))  float  v8f;

#define NB   4096            // batch
#define NF   128             // features
#define NA   8               // attention heads
#define NU   10              // real units
#define UP   16              // padded units (one WMMA tile)
#define NGRP (NF*NA)         // 1024 (j,a) groups
#define NN1  (NGRP*UP)       // 16384 = padded inner dim

// ---------------------------------------------------------------------------
__global__ __launch_bounds__(256) void prep_k1(const float* __restrict__ kern,
                                               __bf16* __restrict__ k1t,
                                               float* __restrict__ kdiag) {
  int idx = blockIdx.x * 256 + threadIdx.x;      // n*128 + k
  int n = idx >> 7, k = idx & 127;
  int u = n & 15, ja = n >> 4;
  int j = ja >> 3;
  float v = 0.0f;
  if (u < NU) v = kern[(ja * NF + k) * NU + u];
  k1t[idx] = (__bf16)v;
  if (k == j) kdiag[n] = v;                      // zero for padded u
}

// ---------------------------------------------------------------------------
__global__ __launch_bounds__(256) void prep_w2(const float* __restrict__ kern,
                                               const float* __restrict__ fc_w,
                                               __bf16* __restrict__ w2t) {
  int idx = blockIdx.x * 256 + threadIdx.x;      // f*NN1 + n
  int f = idx >> 14;                             // NN1 = 2^14
  int n = idx & (NN1 - 1);
  int u = n & 15, ja = n >> 4, a = ja & 7;
  float v = 0.0f;
  if (u < NU) v = __expf(2.0f * kern[(ja * NF + f) * NU + u]) * fc_w[a] * (1.0f / NF);
  w2t[idx] = (__bf16)v;
}

// ---------------------------------------------------------------------------
// Fused main kernel: one block = 16 batch rows, 8 waves split the 1024 groups.
// ---------------------------------------------------------------------------
__global__ __launch_bounds__(256) void fused_ife(
    const float* __restrict__ x,     const __bf16* __restrict__ k1t,
    const __bf16* __restrict__ w2t,  const float* __restrict__ kdiag,
    const float* __restrict__ fc_b,  const float* __restrict__ gamma,
    const float* __restrict__ beta,  const float* __restrict__ mmean,
    const float* __restrict__ mvar,  float* __restrict__ out) {
  __shared__ float xs[16][128];         // X tile (f32), feeds A1 + corrections
  __shared__ float ztf[8][16][36];      // per-wave raw-Z transpose staging (f32)
  __shared__ float ysum[16][128];       // block Y accumulator

  const int tid  = threadIdx.x;
  const int lane = tid & 31;
  const int w    = tid >> 5;            // wave id 0..7
  const int l16  = lane & 15;
  const int hi   = lane >> 4;           // half-wave select (0/1)
  const int m0   = blockIdx.x * 16;     // batch-row tile base

  // ---- load X tile to LDS ----
  #pragma unroll
  for (int i = tid; i < 16 * 128; i += 256)
    xs[i >> 7][i & 127] = x[(size_t)(m0 + (i >> 7)) * NF + (i & 127)];
  __syncthreads();

  // ---- A1 fragments: Xb rows, K=128 -> 4 k-steps of 32 (bf16) ----
  v16bf a1[4];
  #pragma unroll
  for (int kk = 0; kk < 4; ++kk) {
    int base = kk * 32 + hi * 8;
    v16bf t;
    #pragma unroll
    for (int e = 0; e < 8; ++e) t[e]     = (__bf16)xs[l16][base + e];
    #pragma unroll
    for (int e = 0; e < 8; ++e) t[8 + e] = (__bf16)xs[l16][base + 16 + e];
    a1[kk] = t;
  }

  v8f yacc[8];
  #pragma unroll
  for (int t = 0; t < 8; ++t)
    #pragma unroll
    for (int e = 0; e < 8; ++e) yacc[t][e] = 0.0f;

  // ---- main loop: this wave's 128 groups in pairs (GEMM2 K-step = 32) ----
  const int g_begin = w * 128;
  for (int gp = 0; gp < 64; ++gp) {
    const int g0 = g_begin + gp * 2;    // pair (g0, g0+1) shares the same j

    // GEMM1: two 16x16 Z tiles, raw f32 scattered to LDS (C->A transpose)
    #pragma unroll
    for (int s = 0; s < 2; ++s) {
      const int gs = g0 + s;
      v8f zc;
      #pragma unroll
      for (int e = 0; e < 8; ++e) zc[e] = 0.0f;
      #pragma unroll
      for (int kk = 0; kk < 4; ++kk) {
        const __bf16* bp = k1t + (size_t)(gs * 16 + l16) * 128 + kk * 32 + hi * 8;
        v8bf blo = *(const v8bf*)bp;
        v8bf bhi = *(const v8bf*)(bp + 16);
        v16bf bf = __builtin_shufflevector(blo, bhi, 0,1,2,3,4,5,6,7,8,9,10,11,12,13,14,15);
        zc = __builtin_amdgcn_wmma_f32_16x16x32_bf16(false, a1[kk], false, bf,
                                                     (short)0, zc, false, false);
      }
      #pragma unroll
      for (int r = 0; r < 8; ++r)
        ztf[w][r + hi * 8][s * 16 + l16] = zc[r];   // transpose scatter (b32 stores)
    }

    // A-layout read: lane owns row m=l16; elems 0..7 = g0 u(hi*8+e),
    //                elems 8..15 = g1 u(hi*8+e). 2x contiguous 8-float reads.
    float z0[8], z1[8];
    {
      const float* rowp = &ztf[w][l16][0];
      const int kb = hi * 8;
      #pragma unroll
      for (int e = 0; e < 8; ++e) z0[e] = rowp[kb + e];
      #pragma unroll
      for (int e = 0; e < 8; ++e) z1[e] = rowp[16 + kb + e];
    }

    // rank-1 correction in A-layout: one x read per pair, vector kdiag loads
    {
      const int   j  = g0 >> 3;
      const float cx = xs[l16][j];
      const float4* kd0 = (const float4*)(kdiag + g0 * 16 + hi * 8);
      const float4* kd1 = (const float4*)(kdiag + g0 * 16 + 16 + hi * 8);
      float4 ka = kd0[0], kb4 = kd0[1], kc = kd1[0], kd4 = kd1[1];
      const float* k0 = (const float*)&ka;  const float* k0b = (const float*)&kb4;
      const float* k1v = (const float*)&kc; const float* k1b = (const float*)&kd4;
      #pragma unroll
      for (int e = 0; e < 4; ++e) { z0[e] -= cx * k0[e];  z0[4+e] -= cx * k0b[e]; }
      #pragma unroll
      for (int e = 0; e < 4; ++e) { z1[e] -= cx * k1v[e]; z1[4+e] -= cx * k1b[e]; }
    }

    // softmax over u: row split across lane pair (L, L+16) -> 1 shfl per reduce.
    // hi=0 lanes hold u=0..7 (all valid); hi=1 lanes hold u=8..15 (e<2 valid).
    float mx0 = -3.0e38f, mx1 = -3.0e38f;
    #pragma unroll
    for (int e = 0; e < 8; ++e) {
      if (!hi || e < 2) { mx0 = fmaxf(mx0, z0[e]); mx1 = fmaxf(mx1, z1[e]); }
    }
    mx0 = fmaxf(mx0, __shfl_xor(mx0, 16, 32));
    mx1 = fmaxf(mx1, __shfl_xor(mx1, 16, 32));
    float e0[8], e1[8], s0 = 0.0f, s1 = 0.0f;
    #pragma unroll
    for (int e = 0; e < 8; ++e) {
      bool ok = (!hi) || (e < 2);
      e0[e] = ok ? __expf(z0[e] - mx0) : 0.0f;  s0 += e0[e];
      e1[e] = ok ? __expf(z1[e] - mx1) : 0.0f;  s1 += e1[e];
    }
    s0 += __shfl_xor(s0, 16, 32);
    s1 += __shfl_xor(s1, 16, 32);
    const float i0 = 1.0f / s0, i1 = 1.0f / s1;

    // pack directly into the A2 fragment (already in A-element order)
    v16bf a2;
    #pragma unroll
    for (int e = 0; e < 8; ++e) { a2[e] = (__bf16)(e0[e] * i0); a2[8+e] = (__bf16)(e1[e] * i1); }

    // GEMM2 accumulate: Y(16x128) += Zs(16x32) @ W2-slice(32x128)
    #pragma unroll
    for (int t = 0; t < 8; ++t) {
      const __bf16* bp = w2t + (size_t)(t * 16 + l16) * NN1 + g0 * 16 + hi * 8;
      v8bf blo = *(const v8bf*)bp;            // K 0..7  -> n = g0*16 + base..
      v8bf bhi = *(const v8bf*)(bp + 16);     // K16..23 -> n = g1*16 + base..
      v16bf bf = __builtin_shufflevector(blo, bhi, 0,1,2,3,4,5,6,7,8,9,10,11,12,13,14,15);
      yacc[t] = __builtin_amdgcn_wmma_f32_16x16x32_bf16(false, a2, false, bf,
                                                        (short)0, yacc[t], false, false);
    }
  }

  // ---- deterministic cross-wave Y reduction (wave-ordered, 8 barriers) ----
  for (int ww = 0; ww < 8; ++ww) {
    if (w == ww) {
      #pragma unroll
      for (int t = 0; t < 8; ++t)
        #pragma unroll
        for (int r = 0; r < 8; ++r) {
          float* p = &ysum[r + hi * 8][t * 16 + l16];
          *p = (ww == 0) ? yacc[t][r] : (*p + yacc[t][r]);
        }
    }
    __syncthreads();
  }

  // ---- BatchNorm (inference) + softmax over f ----
  const float g  = gamma[0] * rsqrtf(mvar[0] + 1e-3f);
  const float sh = (fc_b[0] - mmean[0]) * g + beta[0];
  const int   r  = tid >> 4;        // row 0..15 (16-tid group, within a wave)
  const int   c  = tid & 15;
  float vals[8];
  float mx = -3.0e38f;
  #pragma unroll
  for (int e = 0; e < 8; ++e) {
    float v = ysum[r][c + 16 * e] * g + sh;
    vals[e] = v;
    mx = fmaxf(mx, v);
  }
  #pragma unroll
  for (int off = 8; off; off >>= 1) mx = fmaxf(mx, __shfl_xor(mx, off, 16));
  float den = 0.0f;
  #pragma unroll
  for (int e = 0; e < 8; ++e) { vals[e] = __expf(vals[e] - mx); den += vals[e]; }
  #pragma unroll
  for (int off = 8; off; off >>= 1) den += __shfl_xor(den, off, 16);
  const float inv = 1.0f / den;
  #pragma unroll
  for (int e = 0; e < 8; ++e)
    out[(size_t)(m0 + r) * NF + c + 16 * e] = vals[e] * inv;
}

// ---------------------------------------------------------------------------
extern "C" void kernel_launch(void* const* d_in, const int* in_sizes, int n_in,
                              void* d_out, int out_size, void* d_ws, size_t ws_size,
                              hipStream_t stream) {
  (void)in_sizes; (void)n_in; (void)out_size; (void)ws_size;
  const float* x     = (const float*)d_in[0];
  const float* kern  = (const float*)d_in[1];
  const float* fc_w  = (const float*)d_in[2];
  const float* fc_b  = (const float*)d_in[3];
  const float* gam   = (const float*)d_in[4];
  const float* bet   = (const float*)d_in[5];
  const float* mmean = (const float*)d_in[6];
  const float* mvar  = (const float*)d_in[7];
  float* out = (float*)d_out;

  // workspace layout (~8.1 MB total)
  char*   ws    = (char*)d_ws;
  __bf16* k1t   = (__bf16*)ws;                                   // 4 MB
  __bf16* w2t   = (__bf16*)(ws + (size_t)NN1 * 128 * 2);         // 4 MB
  float*  kdiag = (float*)(ws + 2 * (size_t)NN1 * 128 * 2);      // 64 KB

  prep_k1<<<(NN1 * 128) / 256, 256, 0, stream>>>(kern, k1t, kdiag);
  prep_w2<<<(128 * NN1) / 256, 256, 0, stream>>>(kern, fc_w, w2t);
  fused_ife<<<NB / 16, 256, 0, stream>>>(x, k1t, w2t, kdiag, fc_b, gam, bet,
                                         mmean, mvar, out);
}
